// MultiHeadAttentionLayer_70738111365854
// MI455X (gfx1250) — compile-verified, matrix operand order checked by reading, not run on hardware
//
#include <hip/hip_runtime.h>
#include <hip/hip_bf16.h>

// Problem constants (from reference): B=2, S=2048, D=1024, H=16, M=128, HD=64
constexpr int Bc  = 2;
constexpr int Sc  = 2048;
constexpr int Dc  = 1024;
constexpr int Hc  = 16;
constexpr int Mc  = 128;
constexpr int HDc = 64;
constexpr float EPSc  = 1e-5f;
// scale = sqrt(HD)=8 ; inv = 1/sqrt(scale) = 1/sqrt(8)
#define INV_SCALE 0.35355339059327373f

typedef _Float16 v16h __attribute__((ext_vector_type(16)));
typedef _Float16 v8h  __attribute__((ext_vector_type(8)));
typedef float    v8f  __attribute__((ext_vector_type(8)));

// ---------------------------------------------------------------------------
// Fragment load for V_WMMA_F32_16X16X32_F16 (A: 16x32 MxK, B: 32x16 KxN).
// 16-bit A layout (ISA 7.12.2): lanes 0-15 hold row M=lane, halves 0..7 = K
// kg*8..kg*8+7 (kg = lane>>4), halves 8..15 = K kg*8+16..kg*8+23.
// B (KxN) mirrors with N=lane&15. Both operands here are K-contiguous rows,
// so each fragment is two 128-bit global loads.
// ---------------------------------------------------------------------------
__device__ __forceinline__ v16h load_frag_nt(const _Float16* __restrict__ base,
                                             int row, int ld, int k0, int lane) {
    const int kg = (lane >> 4) & 1;
    const _Float16* p = base + (size_t)row * ld + k0 + kg * 8;
    v8h lo = *(const v8h*)(p);
    v8h hi = *(const v8h*)(p + 16);
    return __builtin_shufflevector(lo, hi, 0, 1, 2, 3, 4, 5, 6, 7,
                                   8, 9, 10, 11, 12, 13, 14, 15);
}

// ---------------------------------------------------------------------------
// Generic NT GEMM: C[row, col] = sum_k A[row,k] * B[col,k]  (lda = ldb = K)
// Block = 128 threads (4 waves); each wave computes a 32(M) x 64(N) strip
// (2 A-fragments x 4 B-fragments = 8 WMMAs per K-step; each B fragment is
// reused twice -> 12 x 128-bit loads per 8 WMMAs).
// MODE epilogues:
//  0: out proj -> f32 d_out, + bias
//  1: Q/K proj -> (acc+bias)*inv, f16 to head layout [B,H,S,HD]
//  2: V proj   -> acc+bias, f16 transposed per head [B,H,HD,S]
//  3: phi_q = exp(acc - qn[row]) -> f16 [B,H,S,M]       (per-(b,h) gemm)
//  4: psi_k = exp(acc)           -> f16 [B,H,M,S] (T)   (per-(b,h) gemm)
//  5: KV    = psi^T @ V          -> f16 [B,H,HD,M] (T)  (per-(b,h) gemm)
//  6: combine: ctx = (acc/M + eps*c[col]) / den[row] -> f16 [B,S,D]
// ---------------------------------------------------------------------------
template <int MODE>
__global__ __launch_bounds__(128)
void gemm_nt(const _Float16* __restrict__ A, const _Float16* __restrict__ B,
             const float* __restrict__ bias, float* __restrict__ outF,
             _Float16* __restrict__ outH, const float* __restrict__ aux1,
             const float* __restrict__ aux2, int K) {
    const int lane = threadIdx.x & 31;
    const int wave = threadIdx.x >> 5;
    const int z    = blockIdx.z;

    const _Float16* Ap = A;
    const _Float16* Bp = B;
    if (MODE == 3 || MODE == 4) {           // A: qs/ks head rows, B: w^T per head
        Ap += (size_t)z * Sc * HDc;
        Bp += (size_t)(z & (Hc - 1)) * Mc * HDc;
    } else if (MODE == 5) {                 // A: psiT [M,S], B: Vt [HD,S]
        Ap += (size_t)z * Mc * Sc;
        Bp += (size_t)z * HDc * Sc;
    } else if (MODE == 6) {                 // A: phi_q [S,M], B: KVt [HD,M]
        Ap += (size_t)z * Sc * Mc;
        Bp += (size_t)z * HDc * Mc;
    }

    const int rowBase = blockIdx.y * 128 + wave * 32;
    const int colBase = blockIdx.x * 64;
    const int arow0   = rowBase + (lane & 15);
    const int arow1   = arow0 + 16;

    v8f acc[2][4] = {{v8f{}, v8f{}, v8f{}, v8f{}}, {v8f{}, v8f{}, v8f{}, v8f{}}};
    for (int k0 = 0; k0 < K; k0 += 32) {
        if (k0 + 64 < K) {  // pull A lines two K-steps ahead into WGP$
            __builtin_prefetch(Ap + (size_t)arow0 * K + k0 + 64, 0, 1);
            __builtin_prefetch(Ap + (size_t)arow1 * K + k0 + 64, 0, 1);
        }
        v16h af0 = load_frag_nt(Ap, arow0, K, k0, lane);
        v16h af1 = load_frag_nt(Ap, arow1, K, k0, lane);
#pragma unroll
        for (int nt = 0; nt < 4; ++nt) {
            v16h bf = load_frag_nt(Bp, colBase + nt * 16 + (lane & 15), K, k0, lane);
            acc[0][nt] = __builtin_amdgcn_wmma_f32_16x16x32_f16(
                false, af0, false, bf, (short)0, acc[0][nt], false, false);
            acc[1][nt] = __builtin_amdgcn_wmma_f32_16x16x32_f16(
                false, af1, false, bf, (short)0, acc[1][nt], false, false);
        }
    }

    // C/D layout: lanes 0-15: M = r, N = lane; lanes 16-31: M = 8+r, N = lane-16
    const int mhi  = (lane >> 4) * 8;
    const int nloc = lane & 15;
#pragma unroll
    for (int mt = 0; mt < 2; ++mt) {
#pragma unroll
        for (int nt = 0; nt < 4; ++nt) {
            const int col = colBase + nt * 16 + nloc;
#pragma unroll
            for (int r = 0; r < 8; ++r) {
                const int row = rowBase + mt * 16 + mhi + r;
                const float v = acc[mt][nt][r];
                if (MODE == 0) {
                    outF[(size_t)row * Dc + col] = v + bias[col];
                } else if (MODE == 1) {
                    const float q = (v + bias[col]) * INV_SCALE;
                    const int b = row >> 11, s = row & (Sc - 1);
                    const int h = col >> 6, d = col & (HDc - 1);
                    outH[((((size_t)b * Hc + h) * Sc) + s) * HDc + d] = (_Float16)q;
                } else if (MODE == 2) {
                    const float q = v + bias[col];
                    const int b = row >> 11, s = row & (Sc - 1);
                    const int h = col >> 6, d = col & (HDc - 1);
                    outH[(((size_t)b * Hc + h) * HDc + d) * Sc + s] = (_Float16)q;
                } else if (MODE == 3) {
                    float q = __expf(v - aux1[(size_t)z * Sc + row]);
                    q = fminf(q, 60000.0f);
                    outH[((size_t)z * Sc + row) * Mc + col] = (_Float16)q;
                } else if (MODE == 4) {
                    float q = fminf(__expf(v), 60000.0f);
                    outH[((size_t)z * Mc + col) * Sc + row] = (_Float16)q;
                } else if (MODE == 5) {
                    outH[(size_t)z * HDc * Mc + (size_t)col * Mc + row] = (_Float16)v;
                } else if (MODE == 6) {
                    const float num = v * (1.0f / Mc) + EPSc * aux2[(size_t)z * HDc + col];
                    const float q   = num / aux1[(size_t)z * Sc + row];
                    const int b = z >> 4, h = z & (Hc - 1);
                    outH[((size_t)b * Sc + row) * Dc + h * HDc + col] = (_Float16)q;
                }
            }
        }
    }
}

// --------------------------- small helper kernels ---------------------------
__global__ void k_cvt_f16(const float* __restrict__ in, _Float16* __restrict__ out, int n) {
    for (int i = blockIdx.x * blockDim.x + threadIdx.x; i < n; i += gridDim.x * blockDim.x)
        out[i] = (_Float16)in[i];
}

// w[H,HD,M] (f32) -> wT[H,M,HD] (f16)
__global__ void k_transpose_w(const float* __restrict__ in, _Float16* __restrict__ out) {
    const int idx = blockIdx.x * blockDim.x + threadIdx.x;
    if (idx >= Hc * Mc * HDc) return;
    const int d = idx % HDc;
    const int m = (idx / HDc) % Mc;
    const int h = idx / (HDc * Mc);
    out[idx] = (_Float16)in[((size_t)h * HDc + d) * Mc + m];
}

// qn[r] = 0.5*|row|^2 over HD ; optionally ekn[r] = exp(qn[r])
__global__ void k_rownorm(const _Float16* __restrict__ X, float* __restrict__ qn,
                          float* __restrict__ ekn) {
    const int r = blockIdx.x * blockDim.x + threadIdx.x;
    if (r >= Bc * Hc * Sc) return;
    const _Float16* p = X + (size_t)r * HDc;
    float s = 0.0f;
#pragma unroll 8
    for (int i = 0; i < HDc; ++i) {
        const float v = (float)p[i];
        s += v * v;
    }
    s *= 0.5f;
    qn[r] = s;
    if (ekn) ekn[r] = __expf(s);
}

// z[row] = sum_t psiT[row, t]    (rows = B*H*M, rowlen = S)
__global__ void k_rowsum_z(const _Float16* __restrict__ psiT, float* __restrict__ zv) {
    __shared__ float sm[256];
    const int row = blockIdx.x;
    const _Float16* p = psiT + (size_t)row * Sc;
    float s = 0.0f;
    for (int t = threadIdx.x; t < Sc; t += 256) s += (float)p[t];
    sm[threadIdx.x] = s;
    __syncthreads();
    for (int off = 128; off; off >>= 1) {
        if ((int)threadIdx.x < off) sm[threadIdx.x] += sm[threadIdx.x + off];
        __syncthreads();
    }
    if (threadIdx.x == 0) zv[row] = sm[0];
}

// c[row] = sum_t ekn[bh, t] * Vt[row, t]   (rows = B*H*HD)
__global__ void k_rowsum_c(const _Float16* __restrict__ Vt, const float* __restrict__ ekn,
                           float* __restrict__ cvec) {
    __shared__ float sm[256];
    const int row = blockIdx.x;
    const int bh  = row >> 6;
    const _Float16* p = Vt + (size_t)row * Sc;
    const float* w = ekn + (size_t)bh * Sc;
    float s = 0.0f;
    for (int t = threadIdx.x; t < Sc; t += 256) s += (float)p[t] * w[t];
    sm[threadIdx.x] = s;
    __syncthreads();
    for (int off = 128; off; off >>= 1) {
        if ((int)threadIdx.x < off) sm[threadIdx.x] += sm[threadIdx.x + off];
        __syncthreads();
    }
    if (threadIdx.x == 0) cvec[row] = sm[0];
}

// Z[bh] = sum_t ekn[bh, t]
__global__ void k_sum_Z(const float* __restrict__ ekn, float* __restrict__ Zs) {
    __shared__ float sm[256];
    const int bh = blockIdx.x;
    const float* w = ekn + (size_t)bh * Sc;
    float s = 0.0f;
    for (int t = threadIdx.x; t < Sc; t += 256) s += w[t];
    sm[threadIdx.x] = s;
    __syncthreads();
    for (int off = 128; off; off >>= 1) {
        if ((int)threadIdx.x < off) sm[threadIdx.x] += sm[threadIdx.x + off];
        __syncthreads();
    }
    if (threadIdx.x == 0) Zs[bh] = sm[0];
}

// den[r] = dot(phi_q[r,:], z[bh,:]) / M + eps * Z[bh]
__global__ void k_den(const _Float16* __restrict__ phiq, const float* __restrict__ zv,
                      const float* __restrict__ Zs, float* __restrict__ den) {
    const int r = blockIdx.x * blockDim.x + threadIdx.x;
    if (r >= Bc * Hc * Sc) return;
    const int bh = r >> 11;
    const _Float16* p = phiq + (size_t)r * Mc;
    const float* zp = zv + (size_t)bh * Mc;
    float s = 0.0f;
#pragma unroll 8
    for (int m = 0; m < Mc; ++m) s += (float)p[m] * zp[m];
    den[r] = s * (1.0f / Mc) + EPSc * Zs[bh];
}

// ---------------------------------------------------------------------------
extern "C" void kernel_launch(void* const* d_in, const int* in_sizes, int n_in,
                              void* d_out, int out_size, void* d_ws, size_t ws_size,
                              hipStream_t stream) {
    const float* query = (const float*)d_in[0];
    const float* key_  = (const float*)d_in[1];
    const float* value = (const float*)d_in[2];
    const float* Wq = (const float*)d_in[3];  const float* bq = (const float*)d_in[4];
    const float* Wk = (const float*)d_in[5];  const float* bk = (const float*)d_in[6];
    const float* Wv = (const float*)d_in[7];  const float* bv = (const float*)d_in[8];
    const float* Wo = (const float*)d_in[9];  const float* bo = (const float*)d_in[10];
    const float* w1 = (const float*)d_in[11]; const float* w2 = (const float*)d_in[12];
    float* out = (float*)d_out;

    // -------- workspace layout --------
    char* w = (char*)d_ws;
    size_t off = 0;
    auto alloc = [&](size_t bytes) -> void* {
        void* p = w + off;
        off = (off + bytes + 255) & ~(size_t)255;
        return p;
    };
    const size_t BS = (size_t)Bc * Sc;                   // 4096
    _Float16* Xq16 = (_Float16*)alloc(BS * Dc * 2);
    _Float16* Xk16 = (_Float16*)alloc(BS * Dc * 2);
    _Float16* Xv16 = (_Float16*)alloc(BS * Dc * 2);
    _Float16* Wq16 = (_Float16*)alloc((size_t)Dc * Dc * 2);
    _Float16* Wk16 = (_Float16*)alloc((size_t)Dc * Dc * 2);
    _Float16* Wv16 = (_Float16*)alloc((size_t)Dc * Dc * 2);
    _Float16* Wo16 = (_Float16*)alloc((size_t)Dc * Dc * 2);
    _Float16* w1t  = (_Float16*)alloc((size_t)Hc * Mc * HDc * 2);
    _Float16* w2t  = (_Float16*)alloc((size_t)Hc * Mc * HDc * 2);
    _Float16* qh   = (_Float16*)alloc(BS * Dc * 2);      // qs  [B,H,S,HD]
    _Float16* kh   = (_Float16*)alloc(BS * Dc * 2);      // ks  [B,H,S,HD]
    _Float16* Vt   = (_Float16*)alloc(BS * Dc * 2);      // V^T [B,H,HD,S]
    float*    qn   = (float*)alloc((size_t)Bc * Hc * Sc * 4);
    float*    kn   = (float*)alloc((size_t)Bc * Hc * Sc * 4);
    float*    ekn  = (float*)alloc((size_t)Bc * Hc * Sc * 4);
    _Float16* phiq = (_Float16*)alloc((size_t)Bc * Hc * Sc * Mc * 2);
    _Float16* psiT = (_Float16*)alloc((size_t)Bc * Hc * Mc * Sc * 2);
    float*    zv   = (float*)alloc((size_t)Bc * Hc * Mc * 4);
    float*    cvec = (float*)alloc((size_t)Bc * Hc * HDc * 4);
    float*    Zs   = (float*)alloc((size_t)Bc * Hc * 4);
    float*    den  = (float*)alloc((size_t)Bc * Hc * Sc * 4);
    _Float16* KVt  = (_Float16*)alloc((size_t)Bc * Hc * HDc * Mc * 2);
    _Float16* ctx  = (_Float16*)alloc(BS * Dc * 2);      // [B,S,D]
    (void)ws_size; (void)in_sizes; (void)n_in; (void)out_size;

    const dim3 blk(256);
    const int nX = (int)(BS * Dc), nW = Dc * Dc;

    // 1) f32 -> f16 conversions
    k_cvt_f16<<<nX / 256, blk, 0, stream>>>(query, Xq16, nX);
    k_cvt_f16<<<nX / 256, blk, 0, stream>>>(key_,  Xk16, nX);
    k_cvt_f16<<<nX / 256, blk, 0, stream>>>(value, Xv16, nX);
    k_cvt_f16<<<nW / 256, blk, 0, stream>>>(Wq, Wq16, nW);
    k_cvt_f16<<<nW / 256, blk, 0, stream>>>(Wk, Wk16, nW);
    k_cvt_f16<<<nW / 256, blk, 0, stream>>>(Wv, Wv16, nW);
    k_cvt_f16<<<nW / 256, blk, 0, stream>>>(Wo, Wo16, nW);
    k_transpose_w<<<(Hc * Mc * HDc) / 256, blk, 0, stream>>>(w1, w1t);
    k_transpose_w<<<(Hc * Mc * HDc) / 256, blk, 0, stream>>>(w2, w2t);

    const dim3 gblk(128);
    // 2) QKV projections (WMMA NT gemm, K=D). N=1024 -> x=16, rows=4096 -> y=32
    gemm_nt<1><<<dim3(16, 32, 1), gblk, 0, stream>>>(Xq16, Wq16, bq, nullptr, qh, nullptr, nullptr, Dc);
    gemm_nt<1><<<dim3(16, 32, 1), gblk, 0, stream>>>(Xk16, Wk16, bk, nullptr, kh, nullptr, nullptr, Dc);
    gemm_nt<2><<<dim3(16, 32, 1), gblk, 0, stream>>>(Xv16, Wv16, bv, nullptr, Vt, nullptr, nullptr, Dc);

    // 3) row norms: qn, kn, e^{kn}
    k_rownorm<<<(Bc * Hc * Sc) / 256, blk, 0, stream>>>(qh, qn, nullptr);
    k_rownorm<<<(Bc * Hc * Sc) / 256, blk, 0, stream>>>(kh, kn, ekn);

    // 4) feature maps (per-(b,h) WMMA gemms, K=HD=64, N=M=128, rows=S -> y=16)
    gemm_nt<3><<<dim3(2, 16, Bc * Hc), gblk, 0, stream>>>(qh, w1t, nullptr, nullptr, phiq, qn, nullptr, HDc);
    gemm_nt<4><<<dim3(2, 16, Bc * Hc), gblk, 0, stream>>>(kh, w2t, nullptr, nullptr, psiT, nullptr, nullptr, HDc);

    // 5) reductions: z[m], c[d], Z
    k_rowsum_z<<<Bc * Hc * Mc, blk, 0, stream>>>(psiT, zv);
    k_rowsum_c<<<Bc * Hc * HDc, blk, 0, stream>>>(Vt, ekn, cvec);
    k_sum_Z<<<Bc * Hc, blk, 0, stream>>>(ekn, Zs);

    // 6) KV = psi^T @ V  (per-(b,h), K=S=2048, rows=128 -> y=1, N=64)
    gemm_nt<5><<<dim3(1, 1, Bc * Hc), gblk, 0, stream>>>(psiT, Vt, nullptr, nullptr, KVt, nullptr, nullptr, Sc);

    // 7) denominators
    k_den<<<(Bc * Hc * Sc) / 256, blk, 0, stream>>>(phiq, zv, Zs, den);

    // 8) combine: ctx = (phi_q @ KV / M + eps*c) / den  (per-(b,h), K=M=128, rows=S -> y=16)
    gemm_nt<6><<<dim3(1, 16, Bc * Hc), gblk, 0, stream>>>(phiq, KVt, nullptr, nullptr, ctx, den, cvec, Mc);

    // 9) output projection -> f32 d_out
    gemm_nt<0><<<dim3(16, 32, 1), gblk, 0, stream>>>(ctx, Wo16, bo, out, nullptr, nullptr, nullptr, Dc);
}
